// EGNet_49134425866995
// MI455X (gfx1250) — matrix operand order repaired and checked by reading, not compiled, not run on hardware
//
#include <hip/hip_runtime.h>
#include <hip/hip_bf16.h>

typedef __attribute__((ext_vector_type(16))) _Float16 v16h;
typedef __attribute__((ext_vector_type(8)))  _Float16 v8h;
typedef __attribute__((ext_vector_type(8)))  float    v8f;
typedef __attribute__((ext_vector_type(4)))  int      v4i;

#define NN    100000
#define EE    1600000
#define GG    64
#define HIDD  128
#define LL    4
#define BDIM  64            // B * FH
#define CWD   96            // H * B * A
#define EPSV  1e-6f
#define WPAD  136           // 128 + 8 halfs padding: 272B row, 16B aligned, few bank conflicts

#if __has_builtin(__builtin_amdgcn_global_load_async_to_lds_b128) && \
    __has_builtin(__builtin_amdgcn_s_wait_asynccnt)
#define HAVE_ASYNC 1
#else
#define HAVE_ASYNC 0
#endif

// ---------------------------------------------------------------- utilities
__global__ void fill_f32(float* __restrict__ p, float v, long n) {
  long i = (long)blockIdx.x * blockDim.x + threadIdx.x;
  if (i < n) p[i] = v;
}

__device__ __forceinline__ void atomicMaxF32(float* addr, float v) {
  // sign-split trick: int-max for non-negative, uint-min for negative floats
  if (v >= 0.0f) atomicMax((int*)addr, __float_as_int(v));
  else           atomicMin((unsigned int*)addr, __float_as_uint(v));
}

// ------------------------------------------------- WMMA GEMM, K = 128 fixed
// C[nrows, ncols] = A[nrows,128] * W[128,ncols] (+bias), f16 in, f32 acc.
// blockDim = (32,4): 4 waves share one 16-row A tile; each wave owns a
// 16-col tile of a 64-col group. A tile async-copied to LDS (f32, contiguous
// 8KB); W slice staged transposed as f16.
__global__ __launch_bounds__(128) void gemm128_wmma(
    const float* __restrict__ A, const float* __restrict__ W,
    const float* __restrict__ bias, float* __restrict__ C, int ncols) {
  __shared__ __align__(16) float    AsF[16 * HIDD];   // 8 KB
  __shared__ __align__(16) _Float16 Wh[64 * WPAD];    // ~17 KB, Wh[c][k]

  const int tid = threadIdx.y * 32 + threadIdx.x;     // 0..127
  const int rtile = blockIdx.x * 16;
  const int c0 = blockIdx.y * 64;                     // column group base

  // ---- stage A tile: rows rtile..rtile+15, contiguous 2048 f32 in global
  const float* Ablk = A + (long)rtile * HIDD;
#if HAVE_ASYNC
  for (int t = tid; t < 512; t += 128) {              // 512 x 16B chunks
    __builtin_amdgcn_global_load_async_to_lds_b128(
        (__attribute__((address_space(1))) v4i*)(Ablk + t * 4),
        (__attribute__((address_space(3))) v4i*)(AsF + t * 4), 0, 0);
  }
#else
  for (int t = tid; t < 16 * HIDD; t += 128) AsF[t] = Ablk[t];
#endif

  // ---- stage W slice transposed as f16: Wh[c][k] = W[k][c0+c]
  for (int t = tid; t < 64 * HIDD; t += 128) {
    const int k = t >> 6;                             // 0..127
    const int c = t & 63;                             // coalesced over c
    const float v = (c0 + c < ncols) ? W[(long)k * ncols + c0 + c] : 0.0f;
    Wh[c * WPAD + k] = (_Float16)v;
  }
#if HAVE_ASYNC
  __builtin_amdgcn_s_wait_asynccnt(0);
#endif
  __syncthreads();

  const int ctile = threadIdx.y * 16;                 // within 64-col group
  if (c0 + ctile < ncols) {
    const int lane = threadIdx.x;
    const int half = lane >> 4;
    const int l16  = lane & 15;
    const int cloc = ctile + l16;                     // local column
    const float* arow = AsF + l16 * HIDD;
    v8f acc = {};
#pragma unroll
    for (int kk = 0; kk < HIDD; kk += 32) {
      // A fragment (16-bit A 16x32 layout): lanes 0-15 K={0..7,16..23}+kk,
      // lanes 16-31 K={8..15,24..31}+kk; packed 2 K per VGPR.
      const int kb0 = kk + half * 8;
      const int kb1 = kk + 16 + half * 8;
      const float4 f0 = *(const float4*)(arow + kb0);
      const float4 f1 = *(const float4*)(arow + kb0 + 4);
      const float4 f2 = *(const float4*)(arow + kb1);
      const float4 f3 = *(const float4*)(arow + kb1 + 4);
      v16h a;
      a[0]  = (_Float16)f0.x; a[1]  = (_Float16)f0.y;
      a[2]  = (_Float16)f0.z; a[3]  = (_Float16)f0.w;
      a[4]  = (_Float16)f1.x; a[5]  = (_Float16)f1.y;
      a[6]  = (_Float16)f1.z; a[7]  = (_Float16)f1.w;
      a[8]  = (_Float16)f2.x; a[9]  = (_Float16)f2.y;
      a[10] = (_Float16)f2.z; a[11] = (_Float16)f2.w;
      a[12] = (_Float16)f3.x; a[13] = (_Float16)f3.y;
      a[14] = (_Float16)f3.z; a[15] = (_Float16)f3.w;
      // B fragment (16-bit B 32x16): same K order as A (lanes 0-15 K={0..7,16..23}+kk, lanes 16-31 K={8..15,24..31}+kk)
      const v8h b0 = *(const v8h*)(Wh + cloc * WPAD + kb0);
      const v8h b1 = *(const v8h*)(Wh + cloc * WPAD + kb1);
      v16h b;
#pragma unroll
      for (int j = 0; j < 8; ++j) { b[j] = b0[j]; b[j + 8] = b1[j]; }
      acc = __builtin_amdgcn_wmma_f32_16x16x32_f16(false, a, false, b,
                                                   (short)0, acc, false, false);
    }
    const float bv = bias ? bias[c0 + cloc] : 0.0f;
    // C/D layout: VGPR i -> M=i (lanes 0-15) / M=i+8 (lanes 16-31)
#pragma unroll
    for (int i = 0; i < 8; ++i) {
      const int m = i + half * 8;
      C[(long)(rtile + m) * ncols + c0 + cloc] = acc[i] + bv;
    }
  }
}

// ----------------------------------------------------- edge scatter (L2-resident)
__global__ void scatter_edges(const int* __restrict__ ei,
                              const float* __restrict__ bases,
                              float* __restrict__ aggS,
                              float* __restrict__ aggM,
                              float* __restrict__ deg) {
  int e = blockIdx.x * blockDim.x + threadIdx.x;
  if (e >= EE) return;
  __builtin_prefetch(ei + e + 8192, 0, 1);          // global_prefetch_b8
  const int src = ei[e];
  const int dst = ei[EE + e];
  const float4* b4 = (const float4*)(bases + (long)src * BDIM);
  float* srow = aggS + (long)dst * BDIM;
  float* mrow = aggM + (long)dst * BDIM;
#pragma unroll
  for (int q = 0; q < 16; ++q) {
    float4 v = b4[q];
    const int f = q * 4;
    atomicAdd(&srow[f + 0], v.x);  atomicMaxF32(&mrow[f + 0], v.x);
    atomicAdd(&srow[f + 1], v.y);  atomicMaxF32(&mrow[f + 1], v.y);
    atomicAdd(&srow[f + 2], v.z);  atomicMaxF32(&mrow[f + 2], v.z);
    atomicAdd(&srow[f + 3], v.w);  atomicMaxF32(&mrow[f + 3], v.w);
  }
  atomicAdd(&deg[dst], 1.0f);
}

// -------------------------------------- per-node head-wise combine (einsum)
__global__ void combine(const float* __restrict__ wq,     // [N,96]
                        const float* __restrict__ aggS,   // [N,64]
                        const float* __restrict__ aggM,   // [N,64]
                        const float* __restrict__ deg,    // [N]
                        const float* __restrict__ conv_b, // [128]
                        float* __restrict__ g) {          // [N,128]
  long i = (long)blockIdx.x * blockDim.x + threadIdx.x;
  if (i >= (long)NN * HIDD) return;
  const int n  = (int)(i >> 7);
  const int cc = (int)(i & 127);
  const int hh = cc >> 4;
  const int f  = cc & 15;
  const float d = fmaxf(deg[n], 1.0f);
  const float* wrow = wq + (long)n * CWD + hh * 12;   // w[n, hh, 0..11]
  const float* srow = aggS + (long)n * BDIM;
  const float* mrow = aggM + (long)n * BDIM;
  float acc = 0.0f;
#pragma unroll
  for (int bb = 0; bb < 4; ++bb) {
    const float sv = srow[bb * 16 + f];
    float mv = mrow[bb * 16 + f];
    mv = (mv > -1e37f) ? mv : 0.0f;                   // empty rows -> 0
    acc += wrow[0 * 4 + bb] * sv;                     // sum aggregator
    acc += wrow[1 * 4 + bb] * (sv / d);               // mean aggregator
    acc += wrow[2 * 4 + bb] * mv;                     // max aggregator
  }
  g[i] = acc + conv_b[cc];
}

// --------------------------- segment reduction exploiting sorted batch ids
// blockDim = 128 (one thread per channel); each block owns 64 consecutive
// nodes; running sum flushed on (wave-uniform) graph change -> ~1 atomic per
// (block, channel) instead of 64.
#define RPB 64
__global__ __launch_bounds__(128) void seg_reduce_rows(
    const float* __restrict__ v, const int* __restrict__ batch,
    float* __restrict__ acc, int squared) {
  const int c = threadIdx.x;                     // channel 0..127
  const long n0 = (long)blockIdx.x * RPB;
  int curg = -1;
  float run = 0.0f;
  for (int r = 0; r < RPB; ++r) {
    const long n = n0 + r;
    if (n >= NN) break;
    const int gph = batch[n];                    // uniform across the block
    if (gph != curg) {
      if (curg >= 0) atomicAdd(&acc[(long)curg * HIDD + c], run);
      curg = gph;
      run = 0.0f;
    }
    const float s = v[n * HIDD + c];
    run += squared ? s * s : s;
  }
  if (curg >= 0) atomicAdd(&acc[(long)curg * HIDD + c], run);
}

__global__ void div_cnt(float* __restrict__ s, const int* __restrict__ npg) {
  int i = blockIdx.x * blockDim.x + threadIdx.x;
  if (i >= GG * HIDD) return;
  s[i] /= fmaxf((float)npg[i >> 7], 1.0f);
}

__global__ void center(float* __restrict__ g, const float* __restrict__ mean,
                       const int* __restrict__ batch, const float* __restrict__ ms) {
  long i = (long)blockIdx.x * blockDim.x + threadIdx.x;
  if (i >= (long)NN * HIDD) return;
  const int n = (int)(i >> 7);
  const int c = (int)(i & 127);
  g[i] -= mean[(long)batch[n] * HIDD + c] * ms[c];
}

__global__ void norm_relu_residual(const float* __restrict__ g, const float* __restrict__ var,
                                   const int* __restrict__ batch,
                                   const float* __restrict__ wgt, const float* __restrict__ bta,
                                   float* __restrict__ h) {
  long i = (long)blockIdx.x * blockDim.x + threadIdx.x;
  if (i >= (long)NN * HIDD) return;
  const int n = (int)(i >> 7);
  const int c = (int)(i & 127);
  const float inv = rsqrtf(var[(long)batch[n] * HIDD + c] + EPSV);
  const float val = wgt[c] * g[i] * inv + bta[c];
  h[i] += fmaxf(val, 0.0f);
}

// ---------------------------------------------------------------- host side
extern "C" void kernel_launch(void* const* d_in, const int* in_sizes, int n_in,
                              void* d_out, int out_size, void* d_ws, size_t ws_size,
                              hipStream_t stream) {
  const float* x       = (const float*)d_in[0];
  const int*   ei      = (const int*)  d_in[1];
  const int*   batch   = (const int*)  d_in[2];
  const int*   npg     = (const int*)  d_in[3];
  const float* lin_w   = (const float*)d_in[4];
  const float* lin_b   = (const float*)d_in[5];
  const float* bases_w = (const float*)d_in[6];   // [L,128,64]
  const float* comb_w  = (const float*)d_in[7];   // [L,128,96]
  const float* comb_b  = (const float*)d_in[8];   // [L,96]
  const float* conv_b  = (const float*)d_in[9];   // [L,128]
  const float* gn_w    = (const float*)d_in[10];
  const float* gn_b    = (const float*)d_in[11];
  const float* gn_ms   = (const float*)d_in[12];

  float* h = (float*)d_out;                        // [N,128]  (output 0)
  float* y = h + (long)NN * HIDD;                  // [G,128]  (output 1)

  float* ws    = (float*)d_ws;
  float* bases = ws;  ws += (long)NN * BDIM;
  float* aggS  = ws;  ws += (long)NN * BDIM;
  float* aggM  = ws;  ws += (long)NN * BDIM;
  float* deg   = ws;  ws += NN;
  float* wq    = ws;  ws += (long)NN * CWD;
  float* g     = ws;  ws += (long)NN * HIDD;
  float* stat1 = ws;  ws += GG * HIDD;             // per-graph mean
  float* stat2 = ws;  ws += GG * HIDD;             // per-graph var

  const dim3 wblk(32, 4);
  const long NC = (long)NN * HIDD;                 // 12.8M node-channels
  const long NB = (long)NN * BDIM;
  const int  T  = 256;
  const int  nodeBlocks = (int)((NC + T - 1) / T);
  const int  segBlocks  = (NN + RPB - 1) / RPB;
  const int  gcBlocks   = (GG * HIDD + T - 1) / T;

  // h = x @ lin_w + lin_b
  gemm128_wmma<<<dim3(NN / 16, 2), wblk, 0, stream>>>(x, lin_w, lin_b, h, HIDD);

  for (int l = 0; l < LL; ++l) {
    // bases = h @ bases_w[l]
    gemm128_wmma<<<dim3(NN / 16, 1), wblk, 0, stream>>>(
        h, bases_w + (long)l * HIDD * BDIM, nullptr, bases, BDIM);
    // reset aggregation buffers
    fill_f32<<<(int)((NB + T - 1) / T), T, 0, stream>>>(aggS, 0.0f, NB);
    fill_f32<<<(int)((NB + T - 1) / T), T, 0, stream>>>(aggM, -__builtin_inff(), NB);
    fill_f32<<<(NN + T - 1) / T, T, 0, stream>>>(deg, 0.0f, NN);
    // edge gather + scatter reduce (sum / max / degree), L2-resident
    scatter_edges<<<(EE + T - 1) / T, T, 0, stream>>>(ei, bases, aggS, aggM, deg);
    // w = h @ comb_w[l] + comb_b[l]
    gemm128_wmma<<<dim3(NN / 16, 2), wblk, 0, stream>>>(
        h, comb_w + (long)l * HIDD * CWD, comb_b + (long)l * CWD, wq, CWD);
    // g = einsum(w, [sum, mean, max]) + conv_b
    combine<<<nodeBlocks, T, 0, stream>>>(wq, aggS, aggM, deg,
                                          conv_b + (long)l * HIDD, g);
    // GraphNorm (sorted-batch segment reductions)
    fill_f32<<<gcBlocks, T, 0, stream>>>(stat1, 0.0f, GG * HIDD);
    fill_f32<<<gcBlocks, T, 0, stream>>>(stat2, 0.0f, GG * HIDD);
    seg_reduce_rows<<<segBlocks, 128, 0, stream>>>(g, batch, stat1, 0);
    div_cnt<<<gcBlocks, T, 0, stream>>>(stat1, npg);
    center<<<nodeBlocks, T, 0, stream>>>(g, stat1, batch, gn_ms + (long)l * HIDD);
    seg_reduce_rows<<<segBlocks, 128, 0, stream>>>(g, batch, stat2, 1);
    div_cnt<<<gcBlocks, T, 0, stream>>>(stat2, npg);
    norm_relu_residual<<<nodeBlocks, T, 0, stream>>>(g, stat2, batch,
                                                     gn_w + (long)l * HIDD,
                                                     gn_b + (long)l * HIDD, h);
  }

  // y = segment_mean(h)
  fill_f32<<<gcBlocks, T, 0, stream>>>(y, 0.0f, GG * HIDD);
  seg_reduce_rows<<<segBlocks, 128, 0, stream>>>(h, batch, y, 0);
  div_cnt<<<gcBlocks, T, 0, stream>>>(y, npg);
}
